// PoMHSABlock_77068893160364
// MI455X (gfx1250) — matrix operand order repaired and checked by reading, not compiled, not run on hardware
//
#include <hip/hip_runtime.h>

typedef __attribute__((ext_vector_type(16))) _Float16 v16h;
typedef __attribute__((ext_vector_type(8)))  float    v8f;

#define B_  4
#define C_  256
#define L_  4096
#define H_  8
#define DK_ 32

union Frag16 { v16h h; unsigned u[8]; };

// A-matrix 16x32 f16 per-lane K index for VGPR v (contiguous order, matches B):
// lanes 0-15: v0-7 -> K=0..15 ; lanes 16-31: v0-7 -> K=16..31.
__device__ __forceinline__ int a_kidx(int v, int half) {
  return half * 16 + 2 * v;
}

__device__ __forceinline__ void async_b128(unsigned lds_addr, const void* gaddr) {
  asm volatile("global_load_async_to_lds_b128 %0, %1, off"
               :: "v"(lds_addr), "v"((unsigned long long)(size_t)gaddr)
               : "memory");
}

__device__ __forceinline__ void wait_async_zero() {
  asm volatile("s_wait_asynccnt 0x0" ::: "memory");
}

// Async-copy one 64(row) x 32(col f16) panel (64B/row) global -> LDS.
// 256 16B chunks, 128 threads x 2 chunks each. Tracked by ASYNCcnt.
__device__ __forceinline__ void stage_panel_async(
    const _Float16* __restrict__ src,  // global: row-major, row stride C_
    unsigned lds_base,                 // LDS byte address of dest buffer
    int tid) {
#pragma unroll
  for (int i = 0; i < 2; ++i) {
    int c = tid + i * 128;
    int row = c >> 2, part = c & 3;
    async_b128(lds_base + (unsigned)(row * 64 + part * 16),
               src + (size_t)row * C_ + part * 8);
  }
}

// ---------------- conversion kernels ----------------

__global__ void cvt_f32_f16(const float* __restrict__ src,
                            _Float16* __restrict__ dst, int n, float scale) {
  int i = blockIdx.x * blockDim.x + threadIdx.x;
  if (i < n) dst[i] = (_Float16)(src[i] * scale);
}

// x [B,C,L] f32 -> xt [B,L,C] f16
__global__ void cvt_x_transpose(const float* __restrict__ x,
                                _Float16* __restrict__ xt, int n) {
  int i = blockIdx.x * blockDim.x + threadIdx.x;
  if (i >= n) return;
  int b   = i / (C_ * L_);
  int rem = i % (C_ * L_);
  int c   = rem / L_;
  int l   = rem % L_;
  xt[((size_t)b * L_ + l) * C_ + c] = (_Float16)x[i];
}

// ------- projection GEMM: 128 threads, 64(M) x 64(N) tile per block --------
// B-panel staged into LDS with async loads, double-buffered, shared by 4 waves.
// VMODE 0: out[((b*H+h)*L + l)*DK + dk]   (Q, K layout [B,H,L,DK])
// VMODE 1: out[((b*H+h)*DK + dk)*L + l]   (V transposed  [B,H,DK,L])
template <int VMODE>
__global__ void __launch_bounds__(128)
proj_gemm(const _Float16* __restrict__ W,    // [C_out, C_in] row-major
          const _Float16* __restrict__ Xt,   // [B, L, C]
          _Float16* __restrict__ Out) {
  __shared__ _Float16 xbuf[2][64 * 32];
  const int tid  = threadIdx.x;
  const int wave = tid >> 5;
  const int lane = tid & 31;
  const int l0 = blockIdx.x * 64;
  const int m0 = blockIdx.y * 64 + wave * 16;
  const int b  = blockIdx.z;
  const int half = lane >> 4;
  const int lm   = lane & 15;

  const _Float16* xtile = Xt + (size_t)b * L_ * C_ + (size_t)l0 * C_;
  const unsigned lb0 = (unsigned)(size_t)&xbuf[0][0];
  const unsigned lb1 = (unsigned)(size_t)&xbuf[1][0];

  stage_panel_async(xtile, lb0, tid);  // kc = 0

  v8f acc[4] = {{}, {}, {}, {}};
  for (int kc = 0; kc < C_ / 32; ++kc) {
    wait_async_zero();
    __syncthreads();  // current panel visible; prior DS reads drained
    if (kc + 1 < C_ / 32)
      stage_panel_async(xtile + (kc + 1) * 32, (kc & 1) ? lb0 : lb1, tid);

    Frag16 a;
    const _Float16* wrow = W + (m0 + lm) * C_ + kc * 32;
#pragma unroll
    for (int v = 0; v < 8; ++v)
      a.u[v] = *(const unsigned*)(wrow + a_kidx(v, half));

    const _Float16* panel = &xbuf[kc & 1][0];
#pragma unroll
    for (int t = 0; t < 4; ++t) {
      Frag16 bm;
      const _Float16* brow = panel + (t * 16 + lm) * 32 + half * 16;
#pragma unroll
      for (int v = 0; v < 8; ++v)
        bm.u[v] = *(const unsigned*)(brow + 2 * v);
      acc[t] = __builtin_amdgcn_wmma_f32_16x16x32_f16(
          false, a.h, false, bm.h, (short)0, acc[t], false, false);
    }
  }
#pragma unroll
  for (int t = 0; t < 4; ++t) {
#pragma unroll
    for (int r = 0; r < 8; ++r) {
      int o = m0 + r + 8 * half;      // output channel
      int l = l0 + t * 16 + lm;
      int h = o >> 5, dk = o & 31;
      size_t idx;
      if (VMODE == 0) idx = (((size_t)(b * H_ + h)) * L_ + l) * DK_ + dk;
      else            idx = (((size_t)(b * H_ + h)) * DK_ + dk) * L_ + l;
      Out[idx] = (_Float16)acc[t][r];
    }
  }
}

// ---------- flash attention: 8 waves/block share async-staged K/V ----------
// Block = 256 threads = 8 q-tiles of 16; K/V chunks (64 keys) staged to LDS
// once per block (8x global traffic reduction), double-buffered via ASYNCcnt.
__global__ void __launch_bounds__(256)
attn_kernel(const _Float16* __restrict__ Q,    // [B,H,L,DK] (pre-scaled by 1/sqrt(DK) via Wq)
            const _Float16* __restrict__ Km,   // [B,H,L,DK]
            const _Float16* __restrict__ Vt,   // [B,H,DK,L]
            _Float16* __restrict__ Aout) {     // [B,L,C] f16 (post-ReLU)
  __shared__ _Float16 kbuf[2][64 * 32];  // [key][dk]
  __shared__ _Float16 vbuf[2][32 * 64];  // [dk][key]
  __shared__ _Float16 plds[8][16 * 64];  // wave-private P scratch
  const int tid  = threadIdx.x;
  const int wave = tid >> 5;
  const int lane = tid & 31;
  const int q0 = (blockIdx.x * 8 + wave) * 16;
  const int h  = blockIdx.y;
  const int b  = blockIdx.z;
  const int bh = b * H_ + h;
  const int half = lane >> 4;
  const int lm   = lane & 15;
  _Float16* myP = plds[wave];

  const unsigned kbl[2] = {(unsigned)(size_t)&kbuf[0][0],
                           (unsigned)(size_t)&kbuf[1][0]};
  const unsigned vbl[2] = {(unsigned)(size_t)&vbuf[0][0],
                           (unsigned)(size_t)&vbuf[1][0]};

  // each of 256 threads: one b128 of K chunk + one b128 of V chunk
  auto stage_kv = [&](int k0, int buf) {
    {
      int row = tid >> 2, part = tid & 3;   // 64 key rows x 4 chunks
      async_b128(kbl[buf] + (unsigned)(row * 64 + part * 16),
                 Km + ((size_t)bh * L_ + k0 + row) * DK_ + part * 8);
    }
    {
      int row = tid >> 3, part = tid & 7;   // 32 dk rows x 8 chunks
      async_b128(vbl[buf] + (unsigned)(row * 128 + part * 16),
                 Vt + ((size_t)bh * DK_ + row) * L_ + k0 + part * 8);
    }
  };

  // Q A-fragment: M=lm (query row), K = dk (contiguous pairs)
  Frag16 qa;
  const _Float16* qrow = Q + ((size_t)bh * L_ + q0 + lm) * DK_;
#pragma unroll
  for (int v = 0; v < 8; ++v)
    qa.u[v] = *(const unsigned*)(qrow + a_kidx(v, half));

  float mrow[8], ps[8];
  v8f O0 = {}, O1 = {};
#pragma unroll
  for (int r = 0; r < 8; ++r) { mrow[r] = -1e30f; ps[r] = 0.f; }

  stage_kv(0, 0);

  for (int k0 = 0; k0 < L_; k0 += 64) {
    const int cur = (k0 >> 6) & 1;
    wait_async_zero();
    __syncthreads();  // chunk visible to all waves; prior chunk reads drained
    if (k0 + 64 < L_) stage_kv(k0 + 64, cur ^ 1);

    // K^T B-fragments from LDS: N=lm (key), K rows = dk = half*16+2v
    const _Float16* kb = kbuf[cur];
    v8f S[4];
    v8f z = {};
#pragma unroll
    for (int j = 0; j < 4; ++j) {
      Frag16 kf;
#pragma unroll
      for (int v = 0; v < 8; ++v)
        kf.u[v] = *(const unsigned*)&kb[(j * 16 + lm) * 32 + half * 16 + 2 * v];
      S[j] = __builtin_amdgcn_wmma_f32_16x16x32_f16(false, qa.h, false, kf.h,
                                                    (short)0, z, false, false);
    }

    // online softmax; C-layout: VGPR r <-> row r+8*half, lane&15 <-> col
#pragma unroll
    for (int r = 0; r < 8; ++r) {
      float s0 = S[0][r], s1 = S[1][r], s2 = S[2][r], s3 = S[3][r];
      float mx = fmaxf(fmaxf(s0, s1), fmaxf(s2, s3));
      mx = fmaxf(mx, __shfl_xor(mx, 1, 32));
      mx = fmaxf(mx, __shfl_xor(mx, 2, 32));
      mx = fmaxf(mx, __shfl_xor(mx, 4, 32));
      mx = fmaxf(mx, __shfl_xor(mx, 8, 32));
      float nm = fmaxf(mrow[r], mx);
      float alpha = __expf(mrow[r] - nm);
      float e0 = __expf(s0 - nm);
      float e1 = __expf(s1 - nm);
      float e2 = __expf(s2 - nm);
      float e3 = __expf(s3 - nm);
      ps[r] = ps[r] * alpha + (e0 + e1) + (e2 + e3);  // per-lane partial sum
      mrow[r] = nm;
      O0[r] *= alpha;
      O1[r] *= alpha;
      int row = r + 8 * half;
      myP[row * 64 + lm]      = (_Float16)e0;
      myP[row * 64 + 16 + lm] = (_Float16)e1;
      myP[row * 64 + 32 + lm] = (_Float16)e2;
      myP[row * 64 + 48 + lm] = (_Float16)e3;
    }
    // myP is wave-private and LDS ops are in-order per wave: no barrier needed.
    Frag16 pa0, pa1;
#pragma unroll
    for (int v = 0; v < 8; ++v) {
      pa0.u[v] = *(const unsigned*)&myP[lm * 64 + a_kidx(v, half)];
      pa1.u[v] = *(const unsigned*)&myP[lm * 64 + 32 + a_kidx(v, half)];
    }
    // V B-fragments from LDS: N=lm -> dk, K rows = key = c*32 + half*16 + 2v
    const _Float16* vb = vbuf[cur];
    Frag16 vf00, vf01, vf10, vf11;
#pragma unroll
    for (int v = 0; v < 8; ++v) {
      int ko = half * 16 + 2 * v;
      vf00.u[v] = *(const unsigned*)&vb[(lm)      * 64 + ko];       // keys 0-31,  dk 0-15
      vf01.u[v] = *(const unsigned*)&vb[(16 + lm) * 64 + ko];       // keys 0-31,  dk 16-31
      vf10.u[v] = *(const unsigned*)&vb[(lm)      * 64 + 32 + ko];  // keys 32-63, dk 0-15
      vf11.u[v] = *(const unsigned*)&vb[(16 + lm) * 64 + 32 + ko];  // keys 32-63, dk 16-31
    }
    O0 = __builtin_amdgcn_wmma_f32_16x16x32_f16(false, pa0.h, false, vf00.h,
                                                (short)0, O0, false, false);
    O1 = __builtin_amdgcn_wmma_f32_16x16x32_f16(false, pa0.h, false, vf01.h,
                                                (short)0, O1, false, false);
    O0 = __builtin_amdgcn_wmma_f32_16x16x32_f16(false, pa1.h, false, vf10.h,
                                                (short)0, O0, false, false);
    O1 = __builtin_amdgcn_wmma_f32_16x16x32_f16(false, pa1.h, false, vf11.h,
                                                (short)0, O1, false, false);
  }

  // final: reduce deferred row-sums across the 16-lane half, normalize, ReLU
#pragma unroll
  for (int r = 0; r < 8; ++r) {
    float l = ps[r];
    l += __shfl_xor(l, 1, 32);
    l += __shfl_xor(l, 2, 32);
    l += __shfl_xor(l, 4, 32);
    l += __shfl_xor(l, 8, 32);
    float inv = 1.0f / l;
    float o0 = fmaxf(O0[r] * inv, 0.f);  // post-attention ReLU
    float o1 = fmaxf(O1[r] * inv, 0.f);
    int row = r + 8 * half;
    size_t base = ((size_t)b * L_ + q0 + row) * C_ + h * DK_;
    Aout[base + lm]      = (_Float16)o0;
    Aout[base + 16 + lm] = (_Float16)o1;
  }
}

// ------- final projection: 64x64 tile/block, async-staged B panel -----------
__global__ void __launch_bounds__(128)
out_gemm(const _Float16* __restrict__ W,     // Wl f16 [C,C]
         const _Float16* __restrict__ Ain,   // aout [B,L,C] f16
         float* __restrict__ Out) {          // [B,C,L] f32
  __shared__ _Float16 abuf[2][64 * 32];
  const int tid  = threadIdx.x;
  const int wave = tid >> 5;
  const int lane = tid & 31;
  const int l0 = blockIdx.x * 64;
  const int m0 = blockIdx.y * 64 + wave * 16;
  const int b  = blockIdx.z;
  const int half = lane >> 4;
  const int lm   = lane & 15;

  const _Float16* atile = Ain + (size_t)b * L_ * C_ + (size_t)l0 * C_;
  const unsigned lb0 = (unsigned)(size_t)&abuf[0][0];
  const unsigned lb1 = (unsigned)(size_t)&abuf[1][0];

  stage_panel_async(atile, lb0, tid);

  v8f acc[4] = {{}, {}, {}, {}};
  for (int kc = 0; kc < C_ / 32; ++kc) {
    wait_async_zero();
    __syncthreads();
    if (kc + 1 < C_ / 32)
      stage_panel_async(atile + (kc + 1) * 32, (kc & 1) ? lb0 : lb1, tid);

    Frag16 a;
    const _Float16* wrow = W + (m0 + lm) * C_ + kc * 32;
#pragma unroll
    for (int v = 0; v < 8; ++v)
      a.u[v] = *(const unsigned*)(wrow + a_kidx(v, half));

    const _Float16* panel = &abuf[kc & 1][0];
#pragma unroll
    for (int t = 0; t < 4; ++t) {
      Frag16 bm;
      const _Float16* brow = panel + (t * 16 + lm) * 32 + half * 16;
#pragma unroll
      for (int v = 0; v < 8; ++v)
        bm.u[v] = *(const unsigned*)(brow + 2 * v);
      acc[t] = __builtin_amdgcn_wmma_f32_16x16x32_f16(
          false, a.h, false, bm.h, (short)0, acc[t], false, false);
    }
  }
#pragma unroll
  for (int t = 0; t < 4; ++t) {
#pragma unroll
    for (int r = 0; r < 8; ++r) {
      int o = m0 + r + 8 * half;
      Out[(size_t)b * C_ * L_ + (size_t)o * L_ + l0 + t * 16 + lm] = acc[t][r];
    }
  }
}

// ---------------- host launcher ----------------

extern "C" void kernel_launch(void* const* d_in, const int* in_sizes, int n_in,
                              void* d_out, int out_size, void* d_ws, size_t ws_size,
                              hipStream_t stream) {
  (void)in_sizes; (void)n_in; (void)out_size; (void)ws_size;
  const float* x  = (const float*)d_in[0];
  const float* Wq = (const float*)d_in[1];
  const float* Wk = (const float*)d_in[2];
  const float* Wv = (const float*)d_in[3];
  const float* Wl = (const float*)d_in[4];
  float* out = (float*)d_out;

  char* ws = (char*)d_ws;
  const size_t xtBytes  = (size_t)B_ * L_ * C_ * sizeof(_Float16);        // 8 MB
  const size_t wBytes   = (size_t)C_ * C_ * sizeof(_Float16);             // 128 KB
  const size_t qkvBytes = (size_t)B_ * H_ * L_ * DK_ * sizeof(_Float16);  // 8 MB

  _Float16* xt  = (_Float16*)ws;              ws += xtBytes;
  _Float16* wqh = (_Float16*)ws;              ws += wBytes;
  _Float16* wkh = (_Float16*)ws;              ws += wBytes;
  _Float16* wvh = (_Float16*)ws;              ws += wBytes;
  _Float16* wlh = (_Float16*)ws;              ws += wBytes;
  _Float16* Qb  = (_Float16*)ws;              ws += qkvBytes;
  _Float16* Kb  = (_Float16*)ws;              ws += qkvBytes;
  _Float16* Vtb = (_Float16*)ws;              ws += qkvBytes;
  _Float16* Ao  = (_Float16*)ws;              ws += qkvBytes;

  const float qscale = 0.17677669529663687f;  // 1/sqrt(DK): folded into Wq
  const int nW = C_ * C_;
  cvt_f32_f16<<<nW / 256, 256, 0, stream>>>(Wq, wqh, nW, qscale);
  cvt_f32_f16<<<nW / 256, 256, 0, stream>>>(Wk, wkh, nW, 1.0f);
  cvt_f32_f16<<<nW / 256, 256, 0, stream>>>(Wv, wvh, nW, 1.0f);
  cvt_f32_f16<<<nW / 256, 256, 0, stream>>>(Wl, wlh, nW, 1.0f);
  const int nX = B_ * C_ * L_;
  cvt_x_transpose<<<nX / 256, 256, 0, stream>>>(x, xt, nX);

  dim3 gg(L_ / 64, C_ / 64, B_);   // 64 x 4 x 4 blocks, 64x64 tile each
  proj_gemm<0><<<gg, 128, 0, stream>>>(wqh, xt, Qb);
  proj_gemm<0><<<gg, 128, 0, stream>>>(wkh, xt, Kb);
  proj_gemm<1><<<gg, 128, 0, stream>>>(wvh, xt, Vtb);

  dim3 ga(L_ / 128, H_, B_);       // 8 waves/block, 128 queries/block
  attn_kernel<<<ga, 256, 0, stream>>>(Qb, Kb, Vtb, Ao);

  out_gemm<<<gg, 128, 0, stream>>>(wlh, Ao, out);
}
